// Attention_25151328485403
// MI455X (gfx1250) — matrix operand order repaired and checked by reading, not compiled, not run on hardware
//
#include <hip/hip_runtime.h>
#include <hip/hip_bf16.h>

// ---------------------------------------------------------------------------
// MI455X (gfx1250) attention layer, bf16 WMMA pipeline.
//   1) convert x -> bf16, transpose-convert W_qkv, W_out -> bf16 (Bt layout)
//   2) QKV GEMM (v_wmma_f32_16x16x32_bf16), writes Q(*scale), K, V^T bf16
//   3) flash attention per (b,h): scores/softmax in f32 regs, P via LDS, P@V
//   4) out GEMM -> fp32 d_out
// ---------------------------------------------------------------------------

typedef __bf16 bf16;
typedef __attribute__((ext_vector_type(16))) __bf16 v16bf;
typedef __attribute__((ext_vector_type(8)))  __bf16 v8bf;
typedef __attribute__((ext_vector_type(8)))  float  v8f;

#define BATCH 2
#define SEQ   2048
#define DIM   1024
#define NHEAD 16
#define DHEAD 64
#define INNER 1024
#define N3    3072
#define QSCALE 0.125f   // 64^-0.5

union V8F { v8f v; float f[8]; };

// A fragment (16x32 bf16, M rows x K): lane holds row lane%16; K split in
// interleaved 8-chunks per lane half: K = (lane/16)*8 + {0..7, 16..23}.
__device__ __forceinline__ v16bf load_a_frag(const bf16* base, int ld, int k0, int lane) {
    int r  = lane & 15;
    int kb = (lane >> 4) << 3;                 // 0 or 8
    const bf16* p = base + (size_t)r * ld + k0 + kb;
    v8bf lo = *reinterpret_cast<const v8bf*>(p);        // K = kb+0..7
    v8bf hi = *reinterpret_cast<const v8bf*>(p + 16);   // K = kb+16..23
    return __builtin_shufflevector(lo, hi, 0,1,2,3,4,5,6,7,8,9,10,11,12,13,14,15);
}

// B fragment (32x16 bf16, K x N), source stored column-major = Bt row-major
// [N][K]: lane holds column lane%16; K split in interleaved 8-chunks per
// lane half, same as A: K = (lane/16)*8 + {0..7, 16..23}.
__device__ __forceinline__ v16bf load_b_frag(const bf16* baseT, int ld, int k0, int lane) {
    int n = lane & 15;
    int g = (lane >> 4) << 3;                  // 0 or 8
    const bf16* p = baseT + (size_t)n * ld + k0 + g;
    v8bf lo = *reinterpret_cast<const v8bf*>(p);        // K = g+0..7
    v8bf hi = *reinterpret_cast<const v8bf*>(p + 16);   // K = g+16..23
    return __builtin_shufflevector(lo, hi, 0,1,2,3,4,5,6,7,8,9,10,11,12,13,14,15);
}

__device__ __forceinline__ v8f wmma_bf16(v16bf a, v16bf b, v8f c) {
    return __builtin_amdgcn_wmma_f32_16x16x32_bf16(false, a, false, b, (short)0, c, false, false);
}

// ------------------------- conversion kernels ------------------------------
__global__ void __launch_bounds__(256) k_cvt_x(const float* __restrict__ x,
                                               bf16* __restrict__ xb, int n) {
    int i = blockIdx.x * 256 + threadIdx.x;
    if (i < n) xb[i] = (bf16)x[i];
}

// W:[K][N] row-major  ->  WT:[N][K] bf16 (Bt layout for WMMA B-frags)
__global__ void __launch_bounds__(256) k_cvt_T(const float* __restrict__ W,
                                               bf16* __restrict__ WT, int K, int N) {
    int i = blockIdx.x * 256 + threadIdx.x;
    if (i < K * N) {
        int n = i / K, k = i % K;
        WT[(size_t)n * K + k] = (bf16)W[(size_t)k * N + n];
    }
}

// ------------------------------ QKV GEMM -----------------------------------
// C[4096,3072] = xb[4096,1024] @ Wqkv. Each wave: 16x64 tile, 32 K-steps.
// Epilogue routes columns: [0,1024)->Q (scaled), [1024,2048)->K, rest->V^T.
__global__ void __launch_bounds__(256) k_gemm_qkv(const bf16* __restrict__ xb,
                                                  const bf16* __restrict__ wT,
                                                  bf16* __restrict__ Qb,
                                                  bf16* __restrict__ Kb,
                                                  bf16* __restrict__ Vt) {
    int lane = threadIdx.x & 31;
    int w  = blockIdx.x * 8 + (threadIdx.x >> 5);
    int m0 = (w / (N3 / 64)) * 16;
    int n0 = (w % (N3 / 64)) * 64;
    V8F acc[4];
    #pragma unroll
    for (int t = 0; t < 4; t++)
        for (int i = 0; i < 8; i++) acc[t].f[i] = 0.f;

    for (int k0 = 0; k0 < DIM; k0 += 32) {
        v16bf a = load_a_frag(xb + (size_t)m0 * DIM, DIM, k0, lane);
        #pragma unroll
        for (int t = 0; t < 4; t++) {
            v16bf b = load_b_frag(wT + (size_t)(n0 + 16 * t) * DIM, DIM, k0, lane);
            acc[t].v = wmma_bf16(a, b, acc[t].v);
        }
    }
    int lh = lane >> 4, ln = lane & 15;
    #pragma unroll
    for (int t = 0; t < 4; t++) {
        int col  = n0 + 16 * t + ln;
        int part = col >> 10;            // 0=q 1=k 2=v (64-wide tiles never straddle)
        int cc   = col & 1023;
        int h    = cc >> 6, d = cc & 63;
        #pragma unroll
        for (int r = 0; r < 8; r++) {
            int row = m0 + r + 8 * lh;   // flattened b*n
            int bb = row / SEQ, sn = row % SEQ;
            size_t bh = (size_t)(bb * NHEAD + h);
            float val = acc[t].f[r];
            if (part == 0)      Qb[(bh * SEQ + sn) * DHEAD + d] = (bf16)(val * QSCALE);
            else if (part == 1) Kb[(bh * SEQ + sn) * DHEAD + d] = (bf16)val;
            else                Vt[(bh * DHEAD + d) * SEQ + sn] = (bf16)val;
        }
    }
}

// --------------------------- flash attention -------------------------------
// grid (SEQ/128, B*H), 8 waves/block, 16 query rows per wave, 64-key blocks.
__global__ void __launch_bounds__(256) k_attn(const bf16* __restrict__ Qb,
                                              const bf16* __restrict__ Kb,
                                              const bf16* __restrict__ Vt,
                                              const float* __restrict__ mask,
                                              bf16* __restrict__ AO) {
    __shared__ __align__(16) bf16 Plds[8 * 16 * 64];   // 16 KB, private 16x64/wave
    int lane = threadIdx.x & 31;
    int wv   = threadIdx.x >> 5;
    int bh   = blockIdx.y;
    int bb   = bh / NHEAD, hh = bh % NHEAD;
    const bf16*  Qp = Qb + (size_t)bh * SEQ * DHEAD;
    const bf16*  Kp = Kb + (size_t)bh * SEQ * DHEAD;
    const bf16*  Vp = Vt + (size_t)bh * DHEAD * SEQ;
    const float* mb = mask + (size_t)bb * SEQ;
    int q0 = blockIdx.x * 128 + wv * 16;
    int lh = lane >> 4, ln = lane & 15;

    v16bf aq0 = load_a_frag(Qp + (size_t)q0 * DHEAD, DHEAD, 0, lane);
    v16bf aq1 = load_a_frag(Qp + (size_t)q0 * DHEAD, DHEAD, 32, lane);

    float qm[8], mi[8], li[8];
    V8F acc[4], s[4];
    #pragma unroll
    for (int r = 0; r < 8; r++) {
        qm[r] = mb[q0 + r + 8 * lh];
        mi[r] = -1e30f; li[r] = 0.f;
    }
    #pragma unroll
    for (int t = 0; t < 4; t++)
        for (int i = 0; i < 8; i++) acc[t].f[i] = 0.f;

    bf16* myP = Plds + wv * (16 * 64);

    for (int kb0 = 0; kb0 < SEQ; kb0 += 64) {
        float km[4];
        // ----- scores S = (Q*scale) @ K^T, 16x64 in 4 C-tiles
        #pragma unroll
        for (int t = 0; t < 4; t++) {
            for (int i = 0; i < 8; i++) s[t].f[i] = 0.f;
            v16bf b0 = load_b_frag(Kp + (size_t)(kb0 + 16 * t) * DHEAD, DHEAD, 0, lane);
            v16bf b1 = load_b_frag(Kp + (size_t)(kb0 + 16 * t) * DHEAD, DHEAD, 32, lane);
            s[t].v = wmma_bf16(aq0, b0, s[t].v);
            s[t].v = wmma_bf16(aq1, b1, s[t].v);
            km[t] = mb[kb0 + 16 * t + ln];
        }
        // ----- masked online softmax (row stats per (VGPR, lane-half) row)
        float mnew[8], fac[8];
        #pragma unroll
        for (int r = 0; r < 8; r++) {
            float rm = -1e30f;
            #pragma unroll
            for (int t = 0; t < 4; t++) {
                bool keep = (qm[r] * km[t]) > 0.5f;
                float se = keep ? s[t].f[r] : -1e9f;
                s[t].f[r] = se;
                rm = fmaxf(rm, se);
            }
            #pragma unroll
            for (int off = 1; off < 16; off <<= 1)      // stays within lane halves
                rm = fmaxf(rm, __shfl_xor(rm, off, 32));
            mnew[r] = fmaxf(mi[r], rm);
            fac[r]  = __expf(mi[r] - mnew[r]);
            mi[r]   = mnew[r];
        }
        #pragma unroll
        for (int r = 0; r < 8; r++) {
            float ps = 0.f;
            #pragma unroll
            for (int t = 0; t < 4; t++) {
                bool keep = (qm[r] * km[t]) > 0.5f;
                float p = keep ? __expf(s[t].f[r] - mnew[r]) : 0.f;  // zero, per reference
                ps += p;
                myP[(r + 8 * lh) * 64 + 16 * t + ln] = (bf16)p;      // C-layout -> LDS
            }
            #pragma unroll
            for (int off = 1; off < 16; off <<= 1)
                ps += __shfl_xor(ps, off, 32);
            li[r] = li[r] * fac[r] + ps;
            #pragma unroll
            for (int t = 0; t < 4; t++) acc[t].f[r] *= fac[r];
        }
        __syncthreads();                    // order LDS P writes before A-frag reads
        v16bf ap0 = load_a_frag(myP, 64, 0, lane);
        v16bf ap1 = load_a_frag(myP, 64, 32, lane);
        // ----- acc += P @ V  (V^T rows = dims -> B-frag pattern)
        #pragma unroll
        for (int t = 0; t < 4; t++) {
            v16bf bv0 = load_b_frag(Vp + (size_t)(16 * t) * SEQ, SEQ, kb0, lane);
            v16bf bv1 = load_b_frag(Vp + (size_t)(16 * t) * SEQ, SEQ, kb0 + 32, lane);
            acc[t].v = wmma_bf16(ap0, bv0, acc[t].v);
            acc[t].v = wmma_bf16(ap1, bv1, acc[t].v);
        }
        __syncthreads();
    }
    // ----- epilogue: normalize; fully-masked rows -> 0 (matches reference)
    #pragma unroll
    for (int t = 0; t < 4; t++) {
        int d = 16 * t + ln;
        #pragma unroll
        for (int r = 0; r < 8; r++) {
            int sn = q0 + r + 8 * lh;
            float o = (li[r] > 0.f) ? acc[t].f[r] / li[r] : 0.f;
            AO[((size_t)(bb * SEQ + sn)) * INNER + hh * DHEAD + d] = (bf16)o;
        }
    }
}

// ------------------------------ out GEMM -----------------------------------
__global__ void __launch_bounds__(256) k_gemm_out(const bf16* __restrict__ AO,
                                                  const bf16* __restrict__ woT,
                                                  float* __restrict__ out) {
    int lane = threadIdx.x & 31;
    int w  = blockIdx.x * 8 + (threadIdx.x >> 5);
    int m0 = (w / (DIM / 64)) * 16;
    int n0 = (w % (DIM / 64)) * 64;
    V8F acc[4];
    #pragma unroll
    for (int t = 0; t < 4; t++)
        for (int i = 0; i < 8; i++) acc[t].f[i] = 0.f;
    for (int k0 = 0; k0 < INNER; k0 += 32) {
        v16bf a = load_a_frag(AO + (size_t)m0 * INNER, INNER, k0, lane);
        #pragma unroll
        for (int t = 0; t < 4; t++) {
            v16bf b = load_b_frag(woT + (size_t)(n0 + 16 * t) * INNER, INNER, k0, lane);
            acc[t].v = wmma_bf16(a, b, acc[t].v);
        }
    }
    int lh = lane >> 4, ln = lane & 15;
    #pragma unroll
    for (int t = 0; t < 4; t++)
        #pragma unroll
        for (int r = 0; r < 8; r++)
            out[(size_t)(m0 + r + 8 * lh) * DIM + n0 + 16 * t + ln] = acc[t].f[r];
}

// ------------------------------ launcher -----------------------------------
extern "C" void kernel_launch(void* const* d_in, const int* in_sizes, int n_in,
                              void* d_out, int out_size, void* d_ws, size_t ws_size,
                              hipStream_t stream) {
    const float* x    = (const float*)d_in[0];
    const float* mask = (const float*)d_in[1];
    const float* Wqkv = (const float*)d_in[2];
    const float* Wout = (const float*)d_in[3];
    float* out = (float*)d_out;

    char* ws = (char*)d_ws;
    size_t o = 0;
    bf16* xb  = (bf16*)(ws + o); o += (size_t)BATCH * SEQ * DIM * 2;          // 8 MB
    bf16* wqT = (bf16*)(ws + o); o += (size_t)N3 * DIM * 2;                   // 6 MB
    bf16* woT = (bf16*)(ws + o); o += (size_t)DIM * DIM * 2;                  // 2 MB
    bf16* Qb  = (bf16*)(ws + o); o += (size_t)BATCH * NHEAD * SEQ * DHEAD * 2;// 8 MB
    bf16* Kb  = (bf16*)(ws + o); o += (size_t)BATCH * NHEAD * SEQ * DHEAD * 2;// 8 MB
    bf16* Vt  = (bf16*)(ws + o); o += (size_t)BATCH * NHEAD * SEQ * DHEAD * 2;// 8 MB
    bf16* AO  = (bf16*)(ws + o); o += (size_t)BATCH * SEQ * INNER * 2;        // 8 MB

    int nx = BATCH * SEQ * DIM;
    k_cvt_x<<<(nx + 255) / 256, 256, 0, stream>>>(x, xb, nx);
    k_cvt_T<<<(DIM * N3 + 255) / 256, 256, 0, stream>>>(Wqkv, wqT, DIM, N3);
    k_cvt_T<<<(DIM * DIM + 255) / 256, 256, 0, stream>>>(Wout, woT, INNER, DIM);

    k_gemm_qkv<<<(BATCH * SEQ / 16) * (N3 / 64) / 8, 256, 0, stream>>>(xb, wqT, Qb, Kb, Vt);
    k_attn<<<dim3(SEQ / 128, BATCH * NHEAD), 256, 0, stream>>>(Qb, Kb, Vt, mask, AO);
    k_gemm_out<<<(BATCH * SEQ / 16) * (DIM / 64) / 8, 256, 0, stream>>>(AO, woT, out);
}
